// Attention_77403900608902
// MI455X (gfx1250) — matrix operand order repaired and checked by reading, not compiled, not run on hardware
//
#include <hip/hip_runtime.h>
#include <cstdint>

// ---------------------------------------------------------------------------
// Attention forward for MI455X (gfx1250).
// GEMMs: v_wmma_f32_16x16x32_bf16, double-buffered LDS tiles filled with the
// CDNA5 async path (global_load_async_to_lds_b128 + s_wait_asynccnt).
// Block tile 128x256, K-chunk 32, 8 waves, wave tile 64x64 (4x4 subtiles).
// ---------------------------------------------------------------------------

#define B_ 16
#define LQ_ 2048
#define LK_ 2048
#define EMB_ 1024

#define LDSS 40   // padded LDS row stride in halves (80B: 16B-aligned, conflict-free)

typedef __attribute__((ext_vector_type(16))) __bf16 bf16x16;
typedef __attribute__((ext_vector_type(8)))  float  floatx8;

union Frag {
    bf16x16 v;
    uint4   q[2];
};

__device__ __forceinline__ uint16_t f2bf(float f) {
    uint32_t u = __float_as_uint(f);
    uint32_t r = (u + 0x7FFFu + ((u >> 16) & 1u)) >> 16;  // round-to-nearest-even
    return (uint16_t)r;
}

// CDNA5 async global->LDS fill (ASYNCcnt-tracked, no VGPR staging).
__device__ __forceinline__ void async_b128(uint32_t ldsAddr, const uint16_t* g) {
    asm volatile("global_load_async_to_lds_b128 %0, %1, off"
                 :: "v"(ldsAddr), "v"(g) : "memory");
}
__device__ __forceinline__ void wait_async0() {
    asm volatile("s_wait_asynccnt 0x0" ::: "memory");
}

// ---------------------------------------------------------------------------
// fp32 -> bf16 conversion, 4 elements / thread
// ---------------------------------------------------------------------------
__global__ __launch_bounds__(256) void k_f32_to_bf16(const float* __restrict__ src,
                                                     uint16_t* __restrict__ dst,
                                                     long long nvec) {
    long long i = (long long)blockIdx.x * blockDim.x + threadIdx.x;
    if (i >= nvec) return;
    float4 f = ((const float4*)src)[i];
    ushort4 h;
    h.x = f2bf(f.x); h.y = f2bf(f.y); h.z = f2bf(f.z); h.w = f2bf(f.w);
    ((ushort4*)dst)[i] = h;
}

// One K=32 WMMA step out of one LDS buffer: 16 ds_load_b128 -> 16 wmma.
__device__ __forceinline__ void mma_chunk(const uint16_t* sAb, const uint16_t* sBb,
                                          int wm0, int wn0, int lane15, int laneHi,
                                          floatx8 (&acc)[4][4]) {
    Frag fa[4], fb[4];
    #pragma unroll
    for (int i = 0; i < 4; ++i) {
        // A 16x32 frag: lanes 0-15 K {0..7,16..23}, lanes 16-31 K {8..15,24..31}
        const uint16_t* p = sAb + (wm0 + i * 16 + lane15) * LDSS + laneHi * 8;
        fa[i].q[0] = *(const uint4*)(p);
        fa[i].q[1] = *(const uint4*)(p + 16);
    }
    #pragma unroll
    for (int j = 0; j < 4; ++j) {
        // B 32x16 frag: lanes 0-15 K {0..7,16..23}, lanes 16-31 K {8..15,24..31} (N = lane&15)
        const uint16_t* p = sBb + (wn0 + j * 16 + lane15) * LDSS + laneHi * 8;
        fb[j].q[0] = *(const uint4*)(p);
        fb[j].q[1] = *(const uint4*)(p + 16);
    }
    #pragma unroll
    for (int i = 0; i < 4; ++i)
        #pragma unroll
        for (int j = 0; j < 4; ++j)
            acc[i][j] = __builtin_amdgcn_wmma_f32_16x16x32_bf16(
                false, fa[i].v, false, fb[j].v,
                (short)0, acc[i][j], false, false);
}

// ---------------------------------------------------------------------------
// Generic NT GEMM:  C[M,N] = scale * (A[M,K] @ Bm[N,K]^T) + bias[n]
// ---------------------------------------------------------------------------
__global__ __launch_bounds__(256) void k_gemm_nt_bf16(
    const uint16_t* __restrict__ A,  long long strideA, int lda,
    const uint16_t* __restrict__ Bm, long long strideB, int ldb,
    float*    __restrict__ Cf, long long strideCf, int ldcf,
    uint16_t* __restrict__ Cb, long long strideCb, int ldcb,
    int K, float scale, const float* __restrict__ bias, int transOut) {

    __shared__ __align__(16) uint16_t sA[2][128 * LDSS];   // 2 x 10,240 B
    __shared__ __align__(16) uint16_t sB[2][256 * LDSS];   // 2 x 20,480 B

    const int tid    = threadIdx.x;
    const int wave   = tid >> 5;
    const int lane   = tid & 31;
    const int lane15 = lane & 15;
    const int laneHi = lane >> 4;           // 0 | 1

    const int bm0 = blockIdx.y * 128;
    const int bn0 = blockIdx.x * 256;
    const int wm0 = (wave >> 2) * 64;       // 0 | 64
    const int wn0 = (wave & 3) * 64;        // 0,64,128,192

    const uint16_t* Ab = A  + (long long)blockIdx.z * strideA;
    const uint16_t* Bb = Bm + (long long)blockIdx.z * strideB;

    floatx8 acc[4][4] = {};

    // ---- per-thread async-fill slots (b128 each) ----
    // A tile: 128 rows x 32 halves = 512 slots -> 2/thread
    // B tile: 256 rows x 32 halves = 1024 slots -> 4/thread
    const uint16_t* gA[2];
    const uint16_t* gB[4];
    uint32_t lA[2][2], lB[2][4];
    #pragma unroll
    for (int i = 0; i < 2; ++i) {
        const int slot = tid + 256 * i;
        const int row  = slot >> 2;
        const int col  = (slot & 3) * 8;                  // halves
        gA[i] = Ab + (size_t)(bm0 + row) * lda + col;
        lA[0][i] = (uint32_t)(uintptr_t)&sA[0][row * LDSS + col];
        lA[1][i] = (uint32_t)(uintptr_t)&sA[1][row * LDSS + col];
    }
    #pragma unroll
    for (int j = 0; j < 4; ++j) {
        const int slot = tid + 256 * j;
        const int row  = slot >> 2;
        const int col  = (slot & 3) * 8;
        gB[j] = Bb + (size_t)(bn0 + row) * ldb + col;
        lB[0][j] = (uint32_t)(uintptr_t)&sB[0][row * LDSS + col];
        lB[1][j] = (uint32_t)(uintptr_t)&sB[1][row * LDSS + col];
    }

    const int NKC = K >> 5;   // K-chunks of 32 (always even here)

    // prologue: fill buffer 0 with chunk 0
    #pragma unroll
    for (int i = 0; i < 2; ++i) async_b128(lA[0][i], gA[i]);
    #pragma unroll
    for (int j = 0; j < 4; ++j) async_b128(lB[0][j], gB[j]);

    for (int kc = 0; kc < NKC; kc += 2) {
        // ---- buffer 0 ready; refill buffer 1 while computing from 0 ----
        wait_async0();
        __syncthreads();
        if (kc + 1 < NKC) {
            const int ko = (kc + 1) * 32;
            #pragma unroll
            for (int i = 0; i < 2; ++i) async_b128(lA[1][i], gA[i] + ko);
            #pragma unroll
            for (int j = 0; j < 4; ++j) async_b128(lB[1][j], gB[j] + ko);
        }
        mma_chunk(&sA[0][0], &sB[0][0], wm0, wn0, lane15, laneHi, acc);

        // ---- buffer 1 ready; refill buffer 0 while computing from 1 ----
        wait_async0();
        __syncthreads();
        if (kc + 2 < NKC) {
            const int ko = (kc + 2) * 32;
            #pragma unroll
            for (int i = 0; i < 2; ++i) async_b128(lA[0][i], gA[i] + ko);
            #pragma unroll
            for (int j = 0; j < 4; ++j) async_b128(lB[0][j], gB[j] + ko);
        }
        mma_chunk(&sA[1][0], &sB[1][0], wm0, wn0, lane15, laneHi, acc);
    }

    // ---- epilogue: C VGPR r -> (M = r + 8*laneHi, N = lane&15) per 16x16 tile
    float*    cf = Cf ? Cf + (long long)blockIdx.z * strideCf : nullptr;
    uint16_t* cb = Cb ? Cb + (long long)blockIdx.z * strideCb : nullptr;

    #pragma unroll
    for (int i = 0; i < 4; ++i) {
        #pragma unroll
        for (int j = 0; j < 4; ++j) {
            const int n = bn0 + wn0 + j * 16 + lane15;
            const float bv = bias ? bias[n] : 0.0f;
            const int mBase = bm0 + wm0 + i * 16 + laneHi * 8;
            #pragma unroll
            for (int r = 0; r < 8; ++r) {
                const int m = mBase + r;
                const float v = acc[i][j][r] * scale + bv;
                if (cf) cf[(size_t)m * ldcf + n] = v;
                if (cb) {
                    if (transOut) cb[(size_t)n * ldcb + m] = f2bf(v);
                    else          cb[(size_t)m * ldcb + n] = f2bf(v);
                }
            }
        }
    }
}

// ---------------------------------------------------------------------------
// Row softmax over Lk=2048: one block (256 threads) per (b,q) row.
// ---------------------------------------------------------------------------
__global__ __launch_bounds__(256) void k_softmax_row(float* __restrict__ scores,
                                                     uint16_t* __restrict__ attb) {
    const long long row = blockIdx.x;
    float* p = scores + row * (long long)LK_;
    const int tid  = threadIdx.x;
    const int lane = tid & 31;
    const int wave = tid >> 5;

    float x[8];
    float m = -3.4e38f;
    #pragma unroll
    for (int i = 0; i < 8; ++i) {
        x[i] = p[tid + i * 256];
        m = fmaxf(m, x[i]);
    }
    #pragma unroll
    for (int off = 16; off >= 1; off >>= 1)
        m = fmaxf(m, __shfl_xor(m, off, 32));

    __shared__ float red[8];
    if (lane == 0) red[wave] = m;
    __syncthreads();
    float bm = red[0];
    #pragma unroll
    for (int w = 1; w < 8; ++w) bm = fmaxf(bm, red[w]);
    __syncthreads();

    float s = 0.0f;
    #pragma unroll
    for (int i = 0; i < 8; ++i) {
        x[i] = __expf(x[i] - bm);
        s += x[i];
    }
    #pragma unroll
    for (int off = 16; off >= 1; off >>= 1)
        s += __shfl_xor(s, off, 32);
    if (lane == 0) red[wave] = s;
    __syncthreads();
    float tot = 0.0f;
    #pragma unroll
    for (int w = 0; w < 8; ++w) tot += red[w];

    const float inv = __frcp_rn(tot);
    uint16_t* q = attb + row * (long long)LK_;
    #pragma unroll
    for (int i = 0; i < 8; ++i) {
        const float a = x[i] * inv;
        p[tid + i * 256] = a;
        q[tid + i * 256] = f2bf(a);
    }
}

// ---------------------------------------------------------------------------
// Host-side orchestration
// ---------------------------------------------------------------------------
extern "C" void kernel_launch(void* const* d_in, const int* in_sizes, int n_in,
                              void* d_out, int out_size, void* d_ws, size_t ws_size,
                              hipStream_t stream) {
    const float* x   = (const float*)d_in[0];   // [B, LK, E]
    const float* ctx = (const float*)d_in[1];   // [B, LQ, E]
    const float* Wq  = (const float*)d_in[2];   // [E, E]
    const float* Wk  = (const float*)d_in[3];
    const float* Wv  = (const float*)d_in[4];
    const float* Wo  = (const float*)d_in[5];
    const float* bo  = (const float*)d_in[6];   // [E]

    float* out = (float*)d_out;                                   // [B,LQ,E]
    float* att = out + (size_t)B_ * LQ_ * EMB_;                   // [B,LQ,LK]

    const size_t SX  = (size_t)B_ * LK_ * EMB_;
    const size_t WSZ = (size_t)EMB_ * EMB_;
    const size_t SA  = (size_t)B_ * LQ_ * LK_;

    uint16_t* ws   = (uint16_t*)d_ws;
    uint16_t* XB   = ws;             // bf16 x
    uint16_t* CB   = XB  + SX;       // bf16 context
    uint16_t* WQB  = CB  + SX;
    uint16_t* WKB  = WQB + WSZ;
    uint16_t* WVB  = WKB + WSZ;
    uint16_t* WOB  = WVB + WSZ;
    uint16_t* QB   = WOB + WSZ;      // bf16 Q  [B,LQ,E]
    uint16_t* KB   = QB  + SX;       // bf16 K  [B,LK,E]
    uint16_t* VTB  = KB  + SX;       // bf16 V^T [B,E,LK]
    uint16_t* ATTB = VTB + SX;       // bf16 att [B,LQ,LK]
    uint16_t* O0B  = ATTB + SA;      // bf16 att@V [B,LQ,E]

    // ---- fp32 -> bf16 conversions ----
    {
        long long nv = (long long)SX / 4;
        dim3 g((unsigned)((nv + 255) / 256));
        k_f32_to_bf16<<<g, 256, 0, stream>>>(x,   XB, nv);
        k_f32_to_bf16<<<g, 256, 0, stream>>>(ctx, CB, nv);
        long long nw = (long long)WSZ / 4;
        dim3 gw((unsigned)((nw + 255) / 256));
        k_f32_to_bf16<<<gw, 256, 0, stream>>>(Wq, WQB, nw);
        k_f32_to_bf16<<<gw, 256, 0, stream>>>(Wk, WKB, nw);
        k_f32_to_bf16<<<gw, 256, 0, stream>>>(Wv, WVB, nw);
        k_f32_to_bf16<<<gw, 256, 0, stream>>>(Wo, WOB, nw);
    }

    const long long sLE = (long long)LQ_ * EMB_;   // per-batch stride [L,E]
    const long long sQK = (long long)LQ_ * LK_;    // per-batch stride [LQ,LK]

    // ---- Q = ctx @ Wq^T  (bf16 out) ----
    k_gemm_nt_bf16<<<dim3(EMB_ / 256, LQ_ / 128, B_), 256, 0, stream>>>(
        CB, sLE, EMB_,  WQB, 0, EMB_,
        nullptr, 0, 0,  QB, sLE, EMB_,
        EMB_, 1.0f, nullptr, 0);

    // ---- K = x @ Wk^T ----
    k_gemm_nt_bf16<<<dim3(EMB_ / 256, LK_ / 128, B_), 256, 0, stream>>>(
        XB, sLE, EMB_,  WKB, 0, EMB_,
        nullptr, 0, 0,  KB, sLE, EMB_,
        EMB_, 1.0f, nullptr, 0);

    // ---- V^T = (x @ Wv^T)^T  (bf16 out, transposed write) ----
    k_gemm_nt_bf16<<<dim3(EMB_ / 256, LK_ / 128, B_), 256, 0, stream>>>(
        XB, sLE, EMB_,  WVB, 0, EMB_,
        nullptr, 0, 0,  VTB, sLE, LK_,
        EMB_, 1.0f, nullptr, 1);

    // ---- scores = (Q @ K^T) / sqrt(E)  -> f32 straight into d_out att region ----
    k_gemm_nt_bf16<<<dim3(LK_ / 256, LQ_ / 128, B_), 256, 0, stream>>>(
        QB, sLE, EMB_,  KB, sLE, EMB_,
        att, sQK, LK_,  nullptr, 0, 0,
        EMB_, 1.0f / 32.0f, nullptr, 0);

    // ---- softmax rows: in-place f32 + bf16 copy ----
    k_softmax_row<<<dim3(B_ * LQ_), 256, 0, stream>>>(att, ATTB);

    // ---- O0 = att @ V  ==  att @ (V^T)^T  (NT with Vt) ----
    k_gemm_nt_bf16<<<dim3(EMB_ / 256, LQ_ / 128, B_), 256, 0, stream>>>(
        ATTB, sQK, LK_,  VTB, sLE, LK_,
        nullptr, 0, 0,   O0B, sLE, EMB_,
        LK_, 1.0f, nullptr, 0);

    // ---- out = O0 @ Wo^T + bo  -> f32 d_out ----
    k_gemm_nt_bf16<<<dim3(EMB_ / 256, LQ_ / 128, B_), 256, 0, stream>>>(
        O0B, sLE, EMB_,  WOB, 0, EMB_,
        out, sLE, EMB_,  nullptr, 0, 0,
        EMB_, 1.0f, bo, 0);
}
